// SimplifiedLLaMAMoE_86904368268076
// MI455X (gfx1250) — compile-verified, matrix operand order checked by reading, not run on hardware
//
#include <hip/hip_runtime.h>
#include <hip/hip_bf16.h>

// ---------------------------------------------------------------------------
// SimplifiedLLaMAMoE for MI455X (gfx1250, wave32, WMMA)
// Top-2 sparse dispatch via on-device compaction, f16 WMMA
// (v_wmma_f32_16x16x32_f16) with f32 accumulate, fused SwiGLU GEMM1,
// row-scaled scatter GEMM2, deterministic final combine.
// Tile staging uses CDNA5 async DMA: global_load_async_to_lds_b128
// (ASYNCcnt) so the k-loop overlaps DMA with WMMA issue.
// ---------------------------------------------------------------------------

typedef _Float16 half_t;
typedef __attribute__((ext_vector_type(16))) _Float16 v16h;
typedef __attribute__((ext_vector_type(8)))  _Float16 v8h;
typedef __attribute__((ext_vector_type(8)))  float    v8f;

#define NTOK   8192          // B*T
#define CDIM   1024
#define EXP    8
#define IDIM   1408
#define TOPK   2
#define TM     128           // block tile M
#define TN     64            // block tile N
#define TK     32            // k-chunk (one wmma_f32_16x16x32_f16 per 16x16 tile)
#define LDK    (TK + 8)      // LDS row stride in halves (bank-conflict pad)
#define SHARED_BASE 17408    // NTOK*TOPK + EXP*TM  (worst-case expert rows, padded)
#define HROWS  (SHARED_BASE + NTOK)   // 25600 total padded rows incl. shared seg
#define ODUMMY (3*NTOK)      // trash output row for padding rows

// Issue one 16B async DMA global->LDS; tracked by ASYNCcnt.
__device__ __forceinline__ void async_b128(const void* gptr, void* lptr) {
  unsigned lds = (unsigned)(uintptr_t)lptr;            // flat LDS addr truncates to offset
  asm volatile("global_load_async_to_lds_b128 %0, %1, off"
               :: "v"(lds), "v"(gptr) : "memory");
}
__device__ __forceinline__ void wait_async0() {
  asm volatile("s_wait_asynccnt 0x0" ::: "memory");
}

// meta[] layout (global ints): [0..7] counts, [8..15] fill cursors, [16..24] offsets
__global__ void init_meta(int* meta) { if (threadIdx.x < 32) meta[threadIdx.x] = 0; }

__global__ void cvt_f32_f16(const float* __restrict__ s, half_t* __restrict__ d, size_t n) {
  for (size_t i = (size_t)blockIdx.x * blockDim.x + threadIdx.x; i < n;
       i += (size_t)gridDim.x * blockDim.x)
    d[i] = (half_t)s[i];
}

// ---------------- Router: logits -> softmax -> top2, atomic expert counts ----
__global__ __launch_bounds__(256) void router_kernel(
    const float* __restrict__ x, const float* __restrict__ Wg,
    int* __restrict__ meta, int* __restrict__ tok_e, float* __restrict__ tok_w) {
  __shared__ float wg[EXP * CDIM];   // 32 KB gate weights cached in LDS
  const int tid = threadIdx.x;
  for (int i = tid; i < EXP * CDIM; i += 256) wg[i] = Wg[i];
  __syncthreads();
  const int wave = tid >> 5, lane = tid & 31;
  const int n = blockIdx.x * 8 + wave;       // one wave32 per token
  float acc[EXP] = {};
  for (int c = lane; c < CDIM; c += 32) {
    const float xv = x[(size_t)n * CDIM + c];
#pragma unroll
    for (int e = 0; e < EXP; ++e) acc[e] += xv * wg[e * CDIM + c];
  }
#pragma unroll
  for (int e = 0; e < EXP; ++e)
    for (int o = 16; o > 0; o >>= 1) acc[e] += __shfl_down(acc[e], o, 32);
  if (lane == 0) {
    float mx = acc[0];
#pragma unroll
    for (int e = 1; e < EXP; ++e) mx = fmaxf(mx, acc[e]);
    float p[EXP], s = 0.f;
#pragma unroll
    for (int e = 0; e < EXP; ++e) { p[e] = __expf(acc[e] - mx); s += p[e]; }
    const float inv = 1.f / s;
#pragma unroll
    for (int e = 0; e < EXP; ++e) p[e] *= inv;
    int i1 = 0; float p1 = p[0];
#pragma unroll
    for (int e = 1; e < EXP; ++e) if (p[e] > p1) { p1 = p[e]; i1 = e; }   // ties -> lower idx
    int i2 = -1; float p2 = -1.f;
#pragma unroll
    for (int e = 0; e < EXP; ++e) if (e != i1 && p[e] > p2) { p2 = p[e]; i2 = e; }
    tok_e[2 * n] = i1;     tok_w[2 * n] = p1;
    tok_e[2 * n + 1] = i2; tok_w[2 * n + 1] = p2;
    atomicAdd(&meta[i1], 1);
    atomicAdd(&meta[i2], 1);
  }
}

// --------- Segment offsets (padded to TM), padding rows, shared-seg rows -----
__global__ void prefix_pad(int* __restrict__ meta, int* __restrict__ tokidx,
                           float* __restrict__ aw, int* __restrict__ orow) {
  const int tid = threadIdx.x;
  if (tid == 0) {
    int acc = 0;
    for (int e = 0; e < EXP; ++e) {
      meta[16 + e] = acc;
      acc += (meta[e] + TM - 1) / TM * TM;
    }
    meta[24] = acc;   // <= SHARED_BASE by construction
  }
  __syncthreads();
  for (int e = 0; e < EXP; ++e) {
    const int base = meta[16 + e], cnt = meta[e];
    const int padded = (cnt + TM - 1) / TM * TM;
    for (int r = cnt + tid; r < padded; r += blockDim.x) {
      tokidx[base + r] = 0; aw[base + r] = 0.f; orow[base + r] = ODUMMY;
    }
  }
  for (int r = tid; r < NTOK; r += blockDim.x) {   // shared expert = segment 8
    tokidx[SHARED_BASE + r] = r;
    aw[SHARED_BASE + r] = 1.f;
    orow[SHARED_BASE + r] = 2 * NTOK + r;
  }
}

__global__ void fill_assign(const int* __restrict__ tok_e, const float* __restrict__ tok_w,
                            int* __restrict__ meta, int* __restrict__ tokidx,
                            float* __restrict__ aw, int* __restrict__ orow) {
  const int n = blockIdx.x * blockDim.x + threadIdx.x;
  if (n >= NTOK) return;
#pragma unroll
  for (int k = 0; k < TOPK; ++k) {
    const int e = tok_e[2 * n + k];
    const int pos = atomicAdd(&meta[8 + e], 1);
    const int r = meta[16 + e] + pos;
    tokidx[r] = n;
    aw[r] = tok_w[2 * n + k];
    orow[r] = 2 * n + k;
  }
}

// ------- LDS fragment load: 16 halves per lane = two ds_load_b128 -----------
__device__ __forceinline__ v16h frag_ld(const half_t* p) {
  v8h lo = *(const v8h*)p;
  v8h hi = *(const v8h*)(p + 16);
  v16h r;
#pragma unroll
  for (int i = 0; i < 8; ++i) { r[i] = lo[i]; r[i + 8] = hi[i]; }
  return r;
}

#define WMMA_F16(A, B, Cacc) \
  __builtin_amdgcn_wmma_f32_16x16x32_f16(false, (A), false, (B), (short)0, (Cacc), false, false)

// ---------------- GEMM1: H = silu(Xg@W1^T) * (Xg@W2^T), gathered A ----------
__global__ __launch_bounds__(256) void gemm1_swiglu(
    const half_t* __restrict__ xh,
    const half_t* __restrict__ w1h, const half_t* __restrict__ w2h,
    const half_t* __restrict__ s1h, const half_t* __restrict__ s2h,
    const int* __restrict__ meta, const int* __restrict__ tokidx,
    half_t* __restrict__ H) {
  __shared__ half_t As[2][TM][LDK];
  __shared__ half_t B1s[2][TN][LDK];
  __shared__ half_t B2s[2][TN][LDK];

  const int e = blockIdx.z;
  int segbase, segrows;
  if (e < EXP) { segbase = meta[16 + e]; segrows = meta[16 + e + 1] - segbase; }
  else         { segbase = SHARED_BASE;  segrows = NTOK; }
  const int m0 = blockIdx.y * TM;
  if (m0 >= segrows) return;
  const int n0 = blockIdx.x * TN;
  const half_t* b1p = (e < EXP) ? (w1h + (size_t)e * IDIM * CDIM) : s1h;
  const half_t* b2p = (e < EXP) ? (w2h + (size_t)e * IDIM * CDIM) : s2h;

  const int tid = threadIdx.x, lane = tid & 31, wave = tid >> 5;
  const int wm = wave >> 1, wn = wave & 1;            // 4x2 waves -> 32x32 per wave
  const int r16 = lane & 15, ksel = (lane >> 4) << 3; // ISA A/B lane layout

  // Staging: A = 128x32 halves (512 x 16B chunks, 2/thread); B1/B2 = 64x32 (1/thread each)
  const int arow0 = tid >> 2,        akq = tid & 3;
  const int arow1 = (tid >> 2) + 64;
  const int tokA0 = tokidx[segbase + m0 + arow0];
  const int tokA1 = tokidx[segbase + m0 + arow1];
  const int brow = tid >> 2, bkq = tid & 3;

  auto issue = [&](int buf, int kc) {   // async DMA tile kc into LDS[buf]
    async_b128(xh + (size_t)tokA0 * CDIM + kc + akq * 8, &As[buf][arow0][akq * 8]);
    async_b128(xh + (size_t)tokA1 * CDIM + kc + akq * 8, &As[buf][arow1][akq * 8]);
    async_b128(b1p + (size_t)(n0 + brow) * CDIM + kc + bkq * 8, &B1s[buf][brow][bkq * 8]);
    async_b128(b2p + (size_t)(n0 + brow) * CDIM + kc + bkq * 8, &B2s[buf][brow][bkq * 8]);
  };

  v8f accG[2][2] = {}; v8f accU[2][2] = {};
  issue(0, 0); wait_async0(); __syncthreads();

  const int KCH = CDIM / TK;   // 32
  for (int kk = 0; kk < KCH; ++kk) {
    const int cur = kk & 1;
    const bool hn = (kk + 1) < KCH;
    if (hn) issue(cur ^ 1, (kk + 1) * TK);   // DMA next tile while computing
    if (kk + 2 < KCH) {   // stream weight tiles through L2 ahead of time
      __builtin_prefetch(b1p + (size_t)(n0 + brow) * CDIM + (kk + 2) * TK, 0, 1);
      __builtin_prefetch(b2p + (size_t)(n0 + brow) * CDIM + (kk + 2) * TK, 0, 1);
    }
    // load all fragments first, then 8 back-to-back WMMAs
    const v16h a0 = frag_ld(&As[cur][wm * 32 + r16][ksel]);
    const v16h a1 = frag_ld(&As[cur][wm * 32 + 16 + r16][ksel]);
    v16h b1[2], b2[2];
#pragma unroll
    for (int nj = 0; nj < 2; ++nj) {
      b1[nj] = frag_ld(&B1s[cur][wn * 32 + nj * 16 + r16][ksel]);
      b2[nj] = frag_ld(&B2s[cur][wn * 32 + nj * 16 + r16][ksel]);
    }
#pragma unroll
    for (int nj = 0; nj < 2; ++nj) {
      accG[0][nj] = WMMA_F16(a0, b1[nj], accG[0][nj]);
      accG[1][nj] = WMMA_F16(a1, b1[nj], accG[1][nj]);
      accU[0][nj] = WMMA_F16(a0, b2[nj], accU[0][nj]);
      accU[1][nj] = WMMA_F16(a1, b2[nj], accU[1][nj]);
    }
    if (hn) wait_async0();   // our DMA into nxt landed in LDS
    __syncthreads();         // everyone's DMA visible; cur free for overwrite
  }

  // SwiGLU epilogue; C/D layout: M = vgpr + 8*lane[4], N = lane&15
#pragma unroll
  for (int mi = 0; mi < 2; ++mi)
#pragma unroll
    for (int nj = 0; nj < 2; ++nj)
#pragma unroll
      for (int v = 0; v < 8; ++v) {
        const float g = accG[mi][nj][v], u = accU[mi][nj][v];
        const float h = (g / (1.f + __expf(-g))) * u;
        const int rr = segbase + m0 + wm * 32 + mi * 16 + ksel + v;
        const int cc = n0 + wn * 32 + nj * 16 + r16;
        H[(size_t)rr * IDIM + cc] = (half_t)h;
      }
}

// ---------------- GEMM2: O[orow] = aw * (H @ W3^T), scattered rows ----------
__global__ __launch_bounds__(256) void gemm2_proj(
    const half_t* __restrict__ H,
    const half_t* __restrict__ w3h, const half_t* __restrict__ s3h,
    const int* __restrict__ meta, const float* __restrict__ aw,
    const int* __restrict__ orow, float* __restrict__ O) {
  __shared__ half_t As[2][TM][LDK];
  __shared__ half_t Bs[2][TN][LDK];

  const int e = blockIdx.z;
  int segbase, segrows;
  if (e < EXP) { segbase = meta[16 + e]; segrows = meta[16 + e + 1] - segbase; }
  else         { segbase = SHARED_BASE;  segrows = NTOK; }
  const int m0 = blockIdx.y * TM;
  if (m0 >= segrows) return;
  const int n0 = blockIdx.x * TN;                 // over CDIM
  const half_t* bp = (e < EXP) ? (w3h + (size_t)e * CDIM * IDIM) : s3h;

  const int tid = threadIdx.x, lane = tid & 31, wave = tid >> 5;
  const int wm = wave >> 1, wn = wave & 1;
  const int r16 = lane & 15, ksel = (lane >> 4) << 3;

  const int arow0 = tid >> 2, akq = tid & 3;
  const int arow1 = (tid >> 2) + 64;
  const int brow = tid >> 2, bkq = tid & 3;

  auto issue = [&](int buf, int kc) {
    async_b128(H + (size_t)(segbase + m0 + arow0) * IDIM + kc + akq * 8, &As[buf][arow0][akq * 8]);
    async_b128(H + (size_t)(segbase + m0 + arow1) * IDIM + kc + akq * 8, &As[buf][arow1][akq * 8]);
    async_b128(bp + (size_t)(n0 + brow) * IDIM + kc + bkq * 8, &Bs[buf][brow][bkq * 8]);
  };

  v8f acc[2][2] = {};
  issue(0, 0); wait_async0(); __syncthreads();

  const int KCH = IDIM / TK;   // 44
  for (int kk = 0; kk < KCH; ++kk) {
    const int cur = kk & 1;
    const bool hn = (kk + 1) < KCH;
    if (hn) issue(cur ^ 1, (kk + 1) * TK);
    if (kk + 2 < KCH)
      __builtin_prefetch(bp + (size_t)(n0 + brow) * IDIM + (kk + 2) * TK, 0, 1);
    const v16h a0 = frag_ld(&As[cur][wm * 32 + r16][ksel]);
    const v16h a1 = frag_ld(&As[cur][wm * 32 + 16 + r16][ksel]);
    v16h b[2];
#pragma unroll
    for (int nj = 0; nj < 2; ++nj)
      b[nj] = frag_ld(&Bs[cur][wn * 32 + nj * 16 + r16][ksel]);
#pragma unroll
    for (int nj = 0; nj < 2; ++nj) {
      acc[0][nj] = WMMA_F16(a0, b[nj], acc[0][nj]);
      acc[1][nj] = WMMA_F16(a1, b[nj], acc[1][nj]);
    }
    if (hn) wait_async0();
    __syncthreads();
  }

#pragma unroll
  for (int mi = 0; mi < 2; ++mi)
#pragma unroll
    for (int v = 0; v < 8; ++v) {
      const int R = segbase + m0 + wm * 32 + mi * 16 + ksel + v;
      const float w = aw[R];
      const int d = orow[R];
#pragma unroll
      for (int nj = 0; nj < 2; ++nj) {
        const int cc = n0 + wn * 32 + nj * 16 + r16;
        O[(size_t)d * CDIM + cc] = acc[mi][nj][v] * w;
      }
    }
}

// ----- Final combine (fixed order -> deterministic): y = e0 + e1 + shared ---
__global__ void combine_kernel(const float* __restrict__ O, float* __restrict__ out) {
  for (size_t i = (size_t)blockIdx.x * blockDim.x + threadIdx.x;
       i < (size_t)NTOK * CDIM; i += (size_t)gridDim.x * blockDim.x) {
    const size_t n = i / CDIM, c = i % CDIM;
    out[i] = O[(2 * n) * CDIM + c] + O[(2 * n + 1) * CDIM + c] +
             O[((size_t)2 * NTOK + n) * CDIM + c];
  }
}

// ---------------------------------------------------------------------------
extern "C" void kernel_launch(void* const* d_in, const int* in_sizes, int n_in,
                              void* d_out, int out_size, void* d_ws, size_t ws_size,
                              hipStream_t stream) {
  const float* x   = (const float*)d_in[0];
  const float* Wg  = (const float*)d_in[1];
  const float* W1  = (const float*)d_in[2];
  const float* W2  = (const float*)d_in[3];
  const float* W3  = (const float*)d_in[4];
  const float* Ws1 = (const float*)d_in[5];
  const float* Ws2 = (const float*)d_in[6];
  const float* Ws3 = (const float*)d_in[7];
  float* out = (float*)d_out;

  // Workspace carve-up (~266 MB total)
  char* p = (char*)d_ws;
  auto carve = [&](size_t bytes) { char* r = p; p += (bytes + 255) & ~(size_t)255; return r; };
  half_t* xh  = (half_t*)carve((size_t)NTOK * CDIM * 2);
  half_t* w1h = (half_t*)carve((size_t)EXP * IDIM * CDIM * 2);
  half_t* w2h = (half_t*)carve((size_t)EXP * IDIM * CDIM * 2);
  half_t* w3h = (half_t*)carve((size_t)EXP * CDIM * IDIM * 2);
  half_t* s1h = (half_t*)carve((size_t)IDIM * CDIM * 2);
  half_t* s2h = (half_t*)carve((size_t)IDIM * CDIM * 2);
  half_t* s3h = (half_t*)carve((size_t)CDIM * IDIM * 2);
  int*   meta   = (int*)carve(1024);
  int*   tok_e  = (int*)carve((size_t)NTOK * TOPK * 4);
  float* tok_w  = (float*)carve((size_t)NTOK * TOPK * 4);
  int*   tokidx = (int*)carve((size_t)HROWS * 4);
  float* aw     = (float*)carve((size_t)HROWS * 4);
  int*   orowb  = (int*)carve((size_t)HROWS * 4);
  half_t* Hbuf  = (half_t*)carve((size_t)HROWS * IDIM * 2);
  float*  Obuf  = (float*)carve(((size_t)3 * NTOK + 1) * CDIM * 4);

  hipLaunchKernelGGL(init_meta, dim3(1), dim3(64), 0, stream, meta);

  hipLaunchKernelGGL(cvt_f32_f16, dim3(4096), dim3(256), 0, stream, x,   xh,  (size_t)NTOK * CDIM);
  hipLaunchKernelGGL(cvt_f32_f16, dim3(4096), dim3(256), 0, stream, W1,  w1h, (size_t)EXP * IDIM * CDIM);
  hipLaunchKernelGGL(cvt_f32_f16, dim3(4096), dim3(256), 0, stream, W2,  w2h, (size_t)EXP * IDIM * CDIM);
  hipLaunchKernelGGL(cvt_f32_f16, dim3(4096), dim3(256), 0, stream, W3,  w3h, (size_t)EXP * CDIM * IDIM);
  hipLaunchKernelGGL(cvt_f32_f16, dim3(1024), dim3(256), 0, stream, Ws1, s1h, (size_t)IDIM * CDIM);
  hipLaunchKernelGGL(cvt_f32_f16, dim3(1024), dim3(256), 0, stream, Ws2, s2h, (size_t)IDIM * CDIM);
  hipLaunchKernelGGL(cvt_f32_f16, dim3(1024), dim3(256), 0, stream, Ws3, s3h, (size_t)CDIM * IDIM);

  hipLaunchKernelGGL(router_kernel, dim3(NTOK / 8), dim3(256), 0, stream, x, Wg, meta, tok_e, tok_w);
  hipLaunchKernelGGL(prefix_pad,    dim3(1),        dim3(256), 0, stream, meta, tokidx, aw, orowb);
  hipLaunchKernelGGL(fill_assign,   dim3(NTOK / 256), dim3(256), 0, stream, tok_e, tok_w, meta, tokidx, aw, orowb);

  // 9 segments (8 experts + shared); inactive M-tiles early-exit on segrows
  hipLaunchKernelGGL(gemm1_swiglu, dim3(IDIM / TN, 64, EXP + 1), dim3(256), 0, stream,
                     xh, w1h, w2h, s1h, s2h, meta, tokidx, Hbuf);
  hipLaunchKernelGGL(gemm2_proj,   dim3(CDIM / TN, 64, EXP + 1), dim3(256), 0, stream,
                     Hbuf, w3h, s3h, meta, aw, orowb, Obuf);

  hipLaunchKernelGGL(combine_kernel, dim3(4096), dim3(256), 0, stream, Obuf, out);
}
